// LongformerSelfAttentionForPegasus_48808008351859
// MI455X (gfx1250) — matrix operand order repaired and checked by reading, not compiled, not run on hardware
//
#include <hip/hip_runtime.h>
#include <hip/hip_bf16.h>

typedef __bf16 bf16_t;
typedef __attribute__((ext_vector_type(16))) __bf16 v16bf;
typedef __attribute__((ext_vector_type(8)))  __bf16 v8bf;
typedef __attribute__((ext_vector_type(8)))  float   v8f;
typedef __attribute__((ext_vector_type(4)))  float   v4f;
typedef __attribute__((ext_vector_type(4)))  int     v4i;

#define SEQ    4096
#define BSZ    2
#define EMB    1024
#define NH     16
#define HD     64
#define WIN    256
#define NCHUNK (SEQ / WIN)   /* 16 */

#if __has_builtin(__builtin_amdgcn_global_load_async_to_lds_b128)
#define HAVE_ASYNC_LDS 1
#else
#define HAVE_ASYNC_LDS 0
#endif

#if HAVE_ASYNC_LDS
#if __has_builtin(__builtin_amdgcn_s_wait_asynccnt)
#define ASYNC_WAIT() __builtin_amdgcn_s_wait_asynccnt(0)
#else
#define ASYNC_WAIT() asm volatile("s_wait_asynccnt 0x0" ::: "memory")
#endif
#else
#define ASYNC_WAIT()
#endif

static __device__ __forceinline__ v8f wmma_bf16(v16bf a, v16bf b, v8f c) {
  // D = A(16x32 bf16) * B(32x16 bf16) + C(16x16 f32)
  return __builtin_amdgcn_wmma_f32_16x16x32_bf16(false, a, false, b, (short)0, c, false, false);
}
static __device__ __forceinline__ v16bf cat16(v8bf lo, v8bf hi) {
  return __builtin_shufflevector(lo, hi, 0,1,2,3,4,5,6,7,8,9,10,11,12,13,14,15);
}
static __device__ __forceinline__ v8bf cvt8(v4f f0, v4f f1) {
  v8bf o;
#pragma unroll
  for (int j = 0; j < 4; ++j) { o[j] = (bf16_t)f0[j]; o[4 + j] = (bf16_t)f1[j]; }
  return o;
}

// ---------------------------------------------------------------------------
// Pack kernels: one-time fp32 -> bf16 conversion.
// ---------------------------------------------------------------------------
__global__ void __launch_bounds__(256)
pack_x_kernel(const float* __restrict__ q, bf16_t* __restrict__ xb) {
  const size_t idx = ((size_t)blockIdx.x * blockDim.x + threadIdx.x) * 8;
  const int m = (int)(idx >> 10);
  const int k = (int)(idx & (EMB - 1));
  const int b = m >> 12;
  const int s = m & (SEQ - 1);
  const float* src = q + ((size_t)s * BSZ + b) * EMB + k;
  *(v8bf*)(xb + idx) = cvt8(*(const v4f*)src, *(const v4f*)(src + 4));
}

__global__ void __launch_bounds__(256)
pack_w_kernel(const float* __restrict__ Wq, const float* __restrict__ Wk,
              const float* __restrict__ Wv, const float* __restrict__ Wo,
              bf16_t* __restrict__ wb) {
  const int z = blockIdx.y;
  const float* src = (z == 0) ? Wq : (z == 1) ? Wk : (z == 2) ? Wv : Wo;
  bf16_t* dst = wb + (size_t)z * EMB * EMB;
  const size_t idx = ((size_t)blockIdx.x * blockDim.x + threadIdx.x) * 8;
  *(v8bf*)(dst + idx) = cvt8(*(const v4f*)(src + idx), *(const v4f*)(src + idx + 4));
}

// ---------------------------------------------------------------------------
// Kernel 1: fused QKV projection, all-bf16 operands.
// grid = (M/64, N/256, 3), block = 256 (8 waves); wave = 32 rows x 64 cols.
// Each B fragment feeds two WMMAs (a0,a1); short live ranges -> no spills.
// ---------------------------------------------------------------------------
__global__ void __launch_bounds__(256)
qkv_proj_kernel(const bf16_t* __restrict__ xb, const bf16_t* __restrict__ wb,
                const float* __restrict__ bq, const float* __restrict__ bk,
                const float* __restrict__ bv,
                bf16_t* __restrict__ qb, bf16_t* __restrict__ kb,
                bf16_t* __restrict__ vb)
{
  const int wave = threadIdx.x >> 5;
  const int lane = threadIdx.x & 31;
  const int half = lane >> 4;
  const int lcol = lane & 15;

  const int mbase = blockIdx.x * 64 + (wave >> 2) * 32;
  const int nbase = blockIdx.y * 256 + (wave & 3) * 64;
  const int z = blockIdx.z;

  const bf16_t* W   = wb + (size_t)z * EMB * EMB;
  const float* bias = (z == 0) ? bq : (z == 1) ? bk : bv;
  bf16_t*      outp = (z == 0) ? qb : (z == 1) ? kb : vb;

  const bf16_t* arow0 = xb + (size_t)(mbase + lcol) * EMB;
  const bf16_t* arow1 = xb + (size_t)(mbase + 16 + lcol) * EMB;
  const bf16_t* wr[4];
#pragma unroll
  for (int nt = 0; nt < 4; ++nt)
    wr[nt] = W + (size_t)(nbase + nt * 16 + lcol) * EMB;

  v8f acc[2][4];
#pragma unroll
  for (int mt = 0; mt < 2; ++mt)
#pragma unroll
    for (int nt = 0; nt < 4; ++nt)
#pragma unroll
      for (int i = 0; i < 8; ++i) acc[mt][nt][i] = 0.0f;

  for (int k0 = 0; k0 < EMB; k0 += 64) {
    if (k0 + 64 < EMB) {                    // global_prefetch_b8 of next K slab
      __builtin_prefetch(arow0 + k0 + 64, 0, 3);
      __builtin_prefetch(arow1 + k0 + 64, 0, 3);
      __builtin_prefetch(wr[0] + k0 + 64, 0, 3);
      __builtin_prefetch(wr[2] + k0 + 64, 0, 3);
    }
#pragma unroll
    for (int u = 0; u < 2; ++u) {
      const int kk = k0 + u * 32;
      v16bf a0 = cat16(*(const v8bf*)(arow0 + kk + half * 8),
                       *(const v8bf*)(arow0 + kk + 16 + half * 8));
      v16bf a1 = cat16(*(const v8bf*)(arow1 + kk + half * 8),
                       *(const v8bf*)(arow1 + kk + 16 + half * 8));
#pragma unroll
      for (int nt = 0; nt < 4; ++nt) {
        v16bf bfr = cat16(*(const v8bf*)(wr[nt] + kk + half * 8),
                          *(const v8bf*)(wr[nt] + kk + 16 + half * 8));
        acc[0][nt] = wmma_bf16(a0, bfr, acc[0][nt]);
        acc[1][nt] = wmma_bf16(a1, bfr, acc[1][nt]);
      }
    }
  }

#pragma unroll
  for (int nt = 0; nt < 4; ++nt) {
    const int n = nbase + nt * 16 + lcol;
    const float bval = bias[n];
    const int h = n >> 6;
    const int d = n & (HD - 1);
#pragma unroll
    for (int mt = 0; mt < 2; ++mt)
#pragma unroll
      for (int r = 0; r < 8; ++r) {         // C/D: row = r + half*8, col = lcol
        const int m  = mbase + mt * 16 + r + half * 8;
        const int bb = m >> 12;
        const int ss = m & (SEQ - 1);
        outp[(((size_t)bb * NH + h) * SEQ + ss) * HD + d] = (bf16_t)(acc[mt][nt][r] + bval);
      }
  }
}

// ---------------------------------------------------------------------------
// Kernel 2: sliding-window flash attention.
// grid = B*H*NCHUNK, block = 512 (16 waves, one 16-row query tile each).
// K tile staged to LDS via gfx1250 async global->LDS; V loaded+transposed.
// ---------------------------------------------------------------------------
__global__ void __launch_bounds__(512)
swin_attn_kernel(const bf16_t* __restrict__ qb, const bf16_t* __restrict__ kb,
                 const bf16_t* __restrict__ vbuf,
                 const unsigned char* __restrict__ kpm,
                 bf16_t* __restrict__ attn)
{
  __shared__ __align__(16) bf16_t Ks[32 * HD];       // [key][d]
  __shared__ __align__(16) bf16_t Vt[HD * 32];       // [d][key]  (transposed)
  __shared__ __align__(16) bf16_t Ps[16][16 * 32];   // per-wave P staging [row][key]

  const int tid  = threadIdx.x;
  const int wave = tid >> 5;
  const int lane = tid & 31;
  const int half = lane >> 4;
  const int lcol = lane & 15;

  const int c = blockIdx.x & (NCHUNK - 1);
  const int h = (blockIdx.x >> 4) & (NH - 1);
  const int b = blockIdx.x >> 8;

  const size_t base = ((size_t)b * NH + h) * SEQ * HD;
  const int q0 = c * WIN + wave * 16;

  const bf16_t* qrow = qb + base + (size_t)(q0 + lcol) * HD;
  v16bf qlo = cat16(*(const v8bf*)(qrow + half * 8),      *(const v8bf*)(qrow + 16 + half * 8));
  v16bf qhi = cat16(*(const v8bf*)(qrow + 32 + half * 8), *(const v8bf*)(qrow + 48 + half * 8));

  float m_run[8], l_run[8];
  v8f o[4];
#pragma unroll
  for (int r = 0; r < 8; ++r) { m_run[r] = -1e30f; l_run[r] = 0.0f; }
#pragma unroll
  for (int nt = 0; nt < 4; ++nt)
#pragma unroll
    for (int r = 0; r < 8; ++r) o[nt][r] = 0.0f;

  for (int it = 0; it < 24; ++it) {
    const int kk = it * 32;
    __syncthreads();
#if HAVE_ASYNC_LDS
    if (tid < 256) {                       // K tile: async global -> LDS (b128)
      const int key  = tid >> 3;
      const int dblk = (tid & 7) * 8;
      int kpos = c * WIN - WIN + kk + key;
      int kcl  = kpos < 0 ? 0 : (kpos >= SEQ ? SEQ - 1 : kpos);
      __builtin_amdgcn_global_load_async_to_lds_b128(
          (__attribute__((address_space(1))) v4i*)(kb + base + (size_t)kcl * HD + dblk),
          (__attribute__((address_space(3))) v4i*)(&Ks[key * HD + dblk]), 0, 0);
    } else {                               // V tile: load + transpose into LDS
      const int t2   = tid - 256;
      const int key  = t2 >> 3;
      const int dblk = (t2 & 7) * 8;
      int kpos = c * WIN - WIN + kk + key;
      int kcl  = kpos < 0 ? 0 : (kpos >= SEQ ? SEQ - 1 : kpos);
      v8bf vv = *(const v8bf*)(vbuf + base + (size_t)kcl * HD + dblk);
#pragma unroll
      for (int j = 0; j < 8; ++j) Vt[(dblk + j) * 32 + key] = vv[j];
    }
    ASYNC_WAIT();
#else
    if (tid < 256) {
      const int key  = tid >> 3;
      const int dblk = (tid & 7) * 8;
      int kpos = c * WIN - WIN + kk + key;
      int kcl  = kpos < 0 ? 0 : (kpos >= SEQ ? SEQ - 1 : kpos);
      *(v8bf*)(&Ks[key * HD + dblk]) = *(const v8bf*)(kb + base + (size_t)kcl * HD + dblk);
      v8bf vv = *(const v8bf*)(vbuf + base + (size_t)kcl * HD + dblk);
#pragma unroll
      for (int j = 0; j < 8; ++j) Vt[(dblk + j) * 32 + key] = vv[j];
    }
#endif
    __syncthreads();

    // QK^T: load all 4 K fragments, then 4 WMMA (interleaved accumulators)
    const bf16_t* k0row = &Ks[lcol * HD];
    const bf16_t* k1row = &Ks[(16 + lcol) * HD];
    v16bf k0lo = cat16(*(const v8bf*)(k0row + half * 8),      *(const v8bf*)(k0row + 16 + half * 8));
    v16bf k0hi = cat16(*(const v8bf*)(k0row + 32 + half * 8), *(const v8bf*)(k0row + 48 + half * 8));
    v16bf k1lo = cat16(*(const v8bf*)(k1row + half * 8),      *(const v8bf*)(k1row + 16 + half * 8));
    v16bf k1hi = cat16(*(const v8bf*)(k1row + 32 + half * 8), *(const v8bf*)(k1row + 48 + half * 8));
    v8f s[2];
#pragma unroll
    for (int t = 0; t < 2; ++t)
#pragma unroll
      for (int i = 0; i < 8; ++i) s[t][i] = 0.0f;
    s[0] = wmma_bf16(qlo, k0lo, s[0]);
    s[1] = wmma_bf16(qlo, k1lo, s[1]);
    s[0] = wmma_bf16(qhi, k0hi, s[0]);
    s[1] = wmma_bf16(qhi, k1hi, s[1]);

    // scale + exact band / validity / pad mask
#pragma unroll
    for (int t = 0; t < 2; ++t) {
      const int kpos = c * WIN - WIN + kk + t * 16 + lcol;
      const int kcl  = kpos < 0 ? 0 : (kpos >= SEQ ? SEQ - 1 : kpos);
      const bool kok = (kpos >= 0) && (kpos < SEQ) && (kpm[(size_t)b * SEQ + kcl] == 0);
#pragma unroll
      for (int r = 0; r < 8; ++r) {
        const int qpos = q0 + r + half * 8;
        const int rel  = kpos - qpos;
        const float sc = s[t][r] * 0.125f;           // 1/sqrt(64)
        s[t][r] = (kok && rel >= -WIN && rel <= WIN) ? sc : -1e9f;
      }
    }

    // online softmax: per-row reductions within each 16-lane half
    float alpha[8];
#pragma unroll
    for (int r = 0; r < 8; ++r) {
      float mt = fmaxf(s[0][r], s[1][r]);
#pragma unroll
      for (int off = 8; off >= 1; off >>= 1) mt = fmaxf(mt, __shfl_xor(mt, off, 32));
      const float mnew = fmaxf(m_run[r], mt);
      alpha[r] = __expf(m_run[r] - mnew);
      m_run[r] = mnew;
      const float p0 = __expf(s[0][r] - mnew);
      const float p1 = __expf(s[1][r] - mnew);
      s[0][r] = p0; s[1][r] = p1;
      float rs = p0 + p1;
#pragma unroll
      for (int off = 8; off >= 1; off >>= 1) rs += __shfl_xor(rs, off, 32);
      l_run[r] = l_run[r] * alpha[r] + rs;
    }
#pragma unroll
    for (int nt = 0; nt < 4; ++nt)
#pragma unroll
      for (int r = 0; r < 8; ++r) o[nt][r] *= alpha[r];

    // P: C-layout -> A-layout through per-wave LDS (same-wave DS ops in order)
#pragma unroll
    for (int t = 0; t < 2; ++t)
#pragma unroll
      for (int r = 0; r < 8; ++r)
        Ps[wave][(r + half * 8) * 32 + t * 16 + lcol] = (bf16_t)s[t][r];

    const bf16_t* prow = &Ps[wave][lcol * 32];
    v16bf pf = cat16(*(const v8bf*)(prow + half * 8), *(const v8bf*)(prow + 16 + half * 8));

    // PV: load all 4 V fragments, then 4 WMMA
    v16bf vf[4];
#pragma unroll
    for (int nt = 0; nt < 4; ++nt) {
      const bf16_t* vrow = &Vt[(nt * 16 + lcol) * 32];
      vf[nt] = cat16(*(const v8bf*)(vrow + half * 8), *(const v8bf*)(vrow + 16 + half * 8));
    }
#pragma unroll
    for (int nt = 0; nt < 4; ++nt) o[nt] = wmma_bf16(pf, vf[nt], o[nt]);
  }

  // normalize and store attention output as bf16 in [B*S, E]
#pragma unroll
  for (int r = 0; r < 8; ++r) {
    const float inv = 1.0f / l_run[r];
    const int m = b * SEQ + q0 + r + half * 8;
#pragma unroll
    for (int nt = 0; nt < 4; ++nt)
      attn[(size_t)m * EMB + h * HD + nt * 16 + lcol] = (bf16_t)(o[nt][r] * inv);
  }
}

// ---------------------------------------------------------------------------
// Kernel 3: output projection (bf16 x bf16 -> f32), out[s,b,n], [S,B,E].
// grid = (M/64, N/256), block = 256; wave = 32 rows x 64 cols.
// ---------------------------------------------------------------------------
__global__ void __launch_bounds__(256)
out_proj_kernel(const bf16_t* __restrict__ attn, const bf16_t* __restrict__ Wob,
                const float* __restrict__ bo, float* __restrict__ out)
{
  const int wave = threadIdx.x >> 5;
  const int lane = threadIdx.x & 31;
  const int half = lane >> 4;
  const int lcol = lane & 15;

  const int mbase = blockIdx.x * 64 + (wave >> 2) * 32;
  const int nbase = blockIdx.y * 256 + (wave & 3) * 64;

  const bf16_t* arow0 = attn + (size_t)(mbase + lcol) * EMB;
  const bf16_t* arow1 = attn + (size_t)(mbase + 16 + lcol) * EMB;
  const bf16_t* wr[4];
#pragma unroll
  for (int nt = 0; nt < 4; ++nt)
    wr[nt] = Wob + (size_t)(nbase + nt * 16 + lcol) * EMB;

  v8f acc[2][4];
#pragma unroll
  for (int mt = 0; mt < 2; ++mt)
#pragma unroll
    for (int nt = 0; nt < 4; ++nt)
#pragma unroll
      for (int i = 0; i < 8; ++i) acc[mt][nt][i] = 0.0f;

  for (int k0 = 0; k0 < EMB; k0 += 64) {
    if (k0 + 64 < EMB) {
      __builtin_prefetch(arow0 + k0 + 64, 0, 3);
      __builtin_prefetch(arow1 + k0 + 64, 0, 3);
      __builtin_prefetch(wr[0] + k0 + 64, 0, 3);
      __builtin_prefetch(wr[2] + k0 + 64, 0, 3);
    }
#pragma unroll
    for (int u = 0; u < 2; ++u) {
      const int kk = k0 + u * 32;
      v16bf a0 = cat16(*(const v8bf*)(arow0 + kk + half * 8),
                       *(const v8bf*)(arow0 + kk + 16 + half * 8));
      v16bf a1 = cat16(*(const v8bf*)(arow1 + kk + half * 8),
                       *(const v8bf*)(arow1 + kk + 16 + half * 8));
#pragma unroll
      for (int nt = 0; nt < 4; ++nt) {
        v16bf bfr = cat16(*(const v8bf*)(wr[nt] + kk + half * 8),
                          *(const v8bf*)(wr[nt] + kk + 16 + half * 8));
        acc[0][nt] = wmma_bf16(a0, bfr, acc[0][nt]);
        acc[1][nt] = wmma_bf16(a1, bfr, acc[1][nt]);
      }
    }
  }

#pragma unroll
  for (int nt = 0; nt < 4; ++nt) {
    const int n = nbase + nt * 16 + lcol;
    const float bval = bo[n];
#pragma unroll
    for (int mt = 0; mt < 2; ++mt)
#pragma unroll
      for (int r = 0; r < 8; ++r) {
        const int m  = mbase + mt * 16 + r + half * 8;
        const int bb = m >> 12;
        const int ss = m & (SEQ - 1);
        out[((size_t)ss * BSZ + bb) * EMB + n] = acc[mt][nt][r] + bval;   // [S,B,E]
      }
  }
}

// ---------------------------------------------------------------------------
extern "C" void kernel_launch(void* const* d_in, const int* in_sizes, int n_in,
                              void* d_out, int out_size, void* d_ws, size_t ws_size,
                              hipStream_t stream) {
  (void)in_sizes; (void)n_in; (void)out_size; (void)ws_size;

  const float* query = (const float*)d_in[0];
  const float* Wq    = (const float*)d_in[1];
  const float* bq    = (const float*)d_in[2];
  const float* Wk    = (const float*)d_in[3];
  const float* bk    = (const float*)d_in[4];
  const float* Wv    = (const float*)d_in[5];
  const float* bv    = (const float*)d_in[6];
  const float* Wo    = (const float*)d_in[7];
  const float* bo    = (const float*)d_in[8];
  const unsigned char* kpm = (const unsigned char*)d_in[9];
  float* out = (float*)d_out;

  const size_t nX   = (size_t)BSZ * SEQ * EMB;       // 8,388,608
  const size_t nW   = (size_t)EMB * EMB;             // 1,048,576
  const size_t nQKV = (size_t)BSZ * NH * SEQ * HD;   // 8,388,608

  bf16_t* xb    = (bf16_t*)d_ws;
  bf16_t* wb    = xb + nX;            // Wq,Wk,Wv,Wo packed bf16 (4 * nW)
  bf16_t* qb    = wb + 4 * nW;
  bf16_t* kb    = qb + nQKV;
  bf16_t* vb    = kb + nQKV;
  bf16_t* attnb = vb + nQKV;          // [B*S, E]

  pack_x_kernel<<<dim3((unsigned)(nX / 8 / 256)), dim3(256), 0, stream>>>(query, xb);
  pack_w_kernel<<<dim3((unsigned)(nW / 8 / 256), 4), dim3(256), 0, stream>>>(Wq, Wk, Wv, Wo, wb);

  dim3 gProj((BSZ * SEQ) / 64, EMB / 256, 3), bProj(256);
  qkv_proj_kernel<<<gProj, bProj, 0, stream>>>(xb, wb, bq, bk, bv, qb, kb, vb);

  dim3 gAttn(BSZ * NH * NCHUNK), bAttn(512);
  swin_attn_kernel<<<gAttn, bAttn, 0, stream>>>(qb, kb, vb, kpm, attnb);

  dim3 gOut((BSZ * SEQ) / 64, EMB / 256), bOut(256);
  out_proj_kernel<<<gOut, bOut, 0, stream>>>(attnb, wb + 3 * nW, bo, out);
}
